// MambaBlock_6768868458520
// MI455X (gfx1250) — matrix operand order repaired and checked by reading, not compiled, not run on hardware
//
#include <hip/hip_runtime.h>
#include <hip/hip_bf16.h>
#include <math.h>

#define B_SZ    4
#define SEQ_L   4096
#define DIM     1024
#define D_STATE 16
#define D_CONV  4
#define D_INNER 2048
#define XZW     (2 * D_INNER)      // 4096
#define M_ROWS  (B_SZ * SEQ_L)     // 16384
#define NCH     64                 // scan chunks
#define CLEN    (SEQ_L / NCH)      // 64 steps per chunk

typedef __bf16 bf16_t;
typedef bf16_t v16bf __attribute__((ext_vector_type(16)));
typedef bf16_t v8bf  __attribute__((ext_vector_type(8)));
typedef float  v8f   __attribute__((ext_vector_type(8)));

__device__ __forceinline__ float sigmoidf_(float x) { return 1.f / (1.f + __expf(-x)); }

#if defined(__has_builtin)
#if __has_builtin(__builtin_amdgcn_tensor_load_to_lds) && \
    __has_builtin(__builtin_amdgcn_s_wait_tensorcnt)
#define USE_TDM 1
#endif
#endif
#ifndef USE_TDM
#define USE_TDM 0
#endif

// ---------------------------------------------------------------- elementwise
__global__ void k_cvt_bf16(const float* __restrict__ in, bf16_t* __restrict__ out, int n) {
    int i = blockIdx.x * blockDim.x + threadIdx.x;
    if (i < n) out[i] = (bf16_t)in[i];
}

// W[K][N] fp32  ->  WT[N][K] bf16
__global__ void k_transpose_bf16(const float* __restrict__ w, bf16_t* __restrict__ wt,
                                 int K, int N) {
    int i = blockIdx.x * blockDim.x + threadIdx.x;
    if (i >= K * N) return;
    int n = i % N, k = i / N;
    wt[(size_t)n * K + k] = (bf16_t)w[(size_t)k * N + n];
}

// ======================================================================
// 128x128 block tile, BK=32, double-buffered LDS, 8 waves, 8 WMMA/wave/stage.
// ======================================================================
#define BM   128
#define BN   128
#define BK   32
#define LDST 40   // padded LDS row stride (elements) = 80B; TDM pad reproduces this

#if USE_TDM
typedef unsigned int u32x4 __attribute__((ext_vector_type(4)));
typedef int          i32x8 __attribute__((ext_vector_type(8)));
typedef int          i32x4 __attribute__((ext_vector_type(4)));

// Issue one 2-D TDM tile load: tile_d0 x tile_d1 (elements, 2B each) from a
// [tensor_d1 rows x tensor_d0 cols] bf16 tensor into LDS at lds_off, with
// hardware padding of 4 DWORDs (8 elems) after every 16 DWORDs (32 elems)
// -> LDS row stride 40 elements == LDST.
__device__ __forceinline__ void tdm_load_2d(const void* gaddr, unsigned lds_off,
                                            unsigned tensor_d0, unsigned tensor_d1,
                                            unsigned long long stride_elems,
                                            unsigned tile_d0, unsigned tile_d1) {
    unsigned long long ga = reinterpret_cast<unsigned long long>(gaddr);
    u32x4 g0;
    g0[0] = 1u;                                           // count=1, user desc
    g0[1] = lds_off;                                      // lds_addr
    g0[2] = (unsigned)(ga & 0xFFFFFFFFu);                 // global_addr[31:0]
    g0[3] = (unsigned)((ga >> 32) & 0x01FFFFFFu) | (2u << 30);  // addr[56:32]|type=2
    i32x8 g1;
    g1[0] = (int)((1u << 16) |        // data_size = 1 -> 2 bytes
                  (1u << 20) |        // pad_enable
                  (3u << 22) |        // pad_interval code 3 = 16 DWORDs
                  (3u << 25));        // pad_amount  code 3 = 4 DWORDs
    g1[1] = (int)((tensor_d0 & 0xFFFFu) << 16);
    g1[2] = (int)(((tensor_d0 >> 16) & 0xFFFFu) | ((tensor_d1 & 0xFFFFu) << 16));
    g1[3] = (int)(((tensor_d1 >> 16) & 0xFFFFu) | ((tile_d0 & 0xFFFFu) << 16));
    g1[4] = (int)(tile_d1 & 0xFFFFu);                     // tile_dim1, tile_dim2=0
    g1[5] = (int)(stride_elems & 0xFFFFFFFFu);            // tensor_dim0_stride lo
    g1[6] = (int)((stride_elems >> 32) & 0xFFFFu);        // stride hi, dim1_stride=0
    g1[7] = 0;
    i32x4 z4 = {0, 0, 0, 0};
    i32x8 z8 = {0, 0, 0, 0, 0, 0, 0, 0};
    // amdgpu-toolchain (clang-23) 6-arg form: (g0, g1, g2, g3, extra, cpol)
    __builtin_amdgcn_tensor_load_to_lds(g0, g1, z4, z4, z8, 0);
}

// TDM-staged GEMM: wave 0 drives the Tensor Data Mover; all waves do ds_load+WMMA.
template <int EPI>
__global__ __launch_bounds__(256)
void k_gemm_tdm_wmma(const bf16_t* __restrict__ A,
                     const bf16_t* __restrict__ BT,
                     float* __restrict__ C,
                     const float* __restrict__ bias,
                     const float* __restrict__ resid,
                     int M, int N, int K) {
    __shared__ bf16_t As[2][BM * LDST];
    __shared__ bf16_t Bs[2][BN * LDST];

    const int tid  = threadIdx.x;
    const int lane = tid & 31;
    const int w    = tid >> 5;     // wave id: branch on this is wave-uniform
    const int wm   = w >> 1;       // 0..3 -> rows wm*32
    const int wn   = w & 1;        // 0..1 -> cols wn*64

    const int nbn = N / BN;
    const int bm  = blockIdx.x / nbn;
    const int bn  = blockIdx.x - bm * nbn;
    const int m0  = bm * BM;
    const int n0  = bn * BN;

    const unsigned ldsA0 = (unsigned)reinterpret_cast<unsigned long long>(&As[0][0]);
    const unsigned ldsA1 = (unsigned)reinterpret_cast<unsigned long long>(&As[1][0]);
    const unsigned ldsB0 = (unsigned)reinterpret_cast<unsigned long long>(&Bs[0][0]);
    const unsigned ldsB1 = (unsigned)reinterpret_cast<unsigned long long>(&Bs[1][0]);

    auto issue = [&](int s, int buf) {
        tdm_load_2d(A + (size_t)m0 * K + (size_t)s * BK, buf ? ldsA1 : ldsA0,
                    (unsigned)K, (unsigned)M, (unsigned long long)K, BK, BM);
        tdm_load_2d(BT + (size_t)n0 * K + (size_t)s * BK, buf ? ldsB1 : ldsB0,
                    (unsigned)K, (unsigned)N, (unsigned long long)K, BK, BN);
    };

    v8f acc[2][4] = {};
    const int kh = lane >> 4;
    const int ml = lane & 15;
    const int nst = K / BK;

    if (w == 0) {                        // wave-level guard: TDM ignores EXEC
        issue(0, 0);
        __builtin_amdgcn_s_wait_tensorcnt(0);
    }
    __syncthreads();

    for (int s = 0; s < nst; ++s) {
        const int cur = s & 1;
        if (w == 0 && s + 1 < nst) issue(s + 1, cur ^ 1);  // DMA overlaps WMMA

        v16bf af[2], bfr[4];
#pragma unroll
        for (int mi = 0; mi < 2; ++mi) {
            const bf16_t* ap = &As[cur][(wm * 32 + mi * 16 + ml) * LDST];
            v8bf lo = *(const v8bf*)(ap + kh * 8);
            v8bf hi = *(const v8bf*)(ap + 16 + kh * 8);
#pragma unroll
            for (int j = 0; j < 8; ++j) { af[mi][j] = lo[j]; af[mi][8 + j] = hi[j]; }
        }
#pragma unroll
        for (int ti = 0; ti < 4; ++ti) {
            const bf16_t* bp = &Bs[cur][(wn * 64 + ti * 16 + ml) * LDST + kh * 8];
            v8bf lo = *(const v8bf*)bp;
            v8bf hi = *(const v8bf*)(bp + 16);
#pragma unroll
            for (int j = 0; j < 8; ++j) { bfr[ti][j] = lo[j]; bfr[ti][8 + j] = hi[j]; }
        }
#pragma unroll
        for (int mi = 0; mi < 2; ++mi)
#pragma unroll
            for (int ti = 0; ti < 4; ++ti)
                acc[mi][ti] = __builtin_amdgcn_wmma_f32_16x16x32_bf16(
                    false, af[mi], false, bfr[ti], (short)0, acc[mi][ti], false, false);

        if (w == 0 && s + 1 < nst) __builtin_amdgcn_s_wait_tensorcnt(0);
        __syncthreads();
    }

    const int rowoff = kh * 8;
#pragma unroll
    for (int mi = 0; mi < 2; ++mi) {
#pragma unroll
        for (int ti = 0; ti < 4; ++ti) {
            const int col = n0 + wn * 64 + ti * 16 + ml;
#pragma unroll
            for (int i = 0; i < 8; ++i) {
                float v = acc[mi][ti][i];
                const int row = m0 + wm * 32 + mi * 16 + rowoff + i;
                size_t idx = (size_t)row * N + col;
                if (EPI == 1) { v += bias[col]; v = (v > 20.f) ? v : log1pf(__expf(v)); }
                else if (EPI == 2) { v += resid[idx]; }
                C[idx] = v;
            }
        }
    }
}
#endif  // USE_TDM

// --------------------- fallback: manual global->reg->LDS staging (round 2)
template <int EPI>
__global__ __launch_bounds__(256)
void k_gemm_lds_wmma(const bf16_t* __restrict__ A,
                     const bf16_t* __restrict__ BT,
                     float* __restrict__ C,
                     const float* __restrict__ bias,
                     const float* __restrict__ resid,
                     int M, int N, int K) {
    __shared__ bf16_t As[2][BM * LDST];
    __shared__ bf16_t Bs[2][BN * LDST];

    const int tid  = threadIdx.x;
    const int lane = tid & 31;
    const int w    = tid >> 5;
    const int wm   = w >> 1;
    const int wn   = w & 1;

    const int nbn = N / BN;
    const int bm  = blockIdx.x / nbn;
    const int bn  = blockIdx.x - bm * nbn;
    const int m0  = bm * BM;
    const int n0  = bn * BN;

    const int sr = tid >> 1;
    const int sc = (tid & 1) * 16;

    const bf16_t* gA = A  + (size_t)(m0 + sr) * K + sc;
    const bf16_t* gB = BT + (size_t)(n0 + sr) * K + sc;

    v8bf ra0, ra1, rb0, rb1;
    auto gload = [&](int k0) {
        ra0 = *(const v8bf*)(gA + k0);
        ra1 = *(const v8bf*)(gA + k0 + 8);
        rb0 = *(const v8bf*)(gB + k0);
        rb1 = *(const v8bf*)(gB + k0 + 8);
    };
    auto sstore = [&](int buf) {
        v8bf* as = (v8bf*)&As[buf][sr * LDST + sc];
        as[0] = ra0; as[1] = ra1;
        v8bf* bs = (v8bf*)&Bs[buf][sr * LDST + sc];
        bs[0] = rb0; bs[1] = rb1;
    };

    v8f acc[2][4] = {};
    const int kh = lane >> 4;
    const int ml = lane & 15;

    gload(0);
    sstore(0);
    __syncthreads();

    const int nst = K / BK;
    for (int s = 0; s < nst; ++s) {
        const int cur = s & 1;
        if (s + 1 < nst) gload((s + 1) * BK);

        v16bf af[2], bfr[4];
#pragma unroll
        for (int mi = 0; mi < 2; ++mi) {
            const bf16_t* ap = &As[cur][(wm * 32 + mi * 16 + ml) * LDST];
            v8bf lo = *(const v8bf*)(ap + kh * 8);
            v8bf hi = *(const v8bf*)(ap + 16 + kh * 8);
#pragma unroll
            for (int j = 0; j < 8; ++j) { af[mi][j] = lo[j]; af[mi][8 + j] = hi[j]; }
        }
#pragma unroll
        for (int ti = 0; ti < 4; ++ti) {
            const bf16_t* bp = &Bs[cur][(wn * 64 + ti * 16 + ml) * LDST + kh * 8];
            v8bf lo = *(const v8bf*)bp;
            v8bf hi = *(const v8bf*)(bp + 16);
#pragma unroll
            for (int j = 0; j < 8; ++j) { bfr[ti][j] = lo[j]; bfr[ti][8 + j] = hi[j]; }
        }
#pragma unroll
        for (int mi = 0; mi < 2; ++mi)
#pragma unroll
            for (int ti = 0; ti < 4; ++ti)
                acc[mi][ti] = __builtin_amdgcn_wmma_f32_16x16x32_bf16(
                    false, af[mi], false, bfr[ti], (short)0, acc[mi][ti], false, false);

        if (s + 1 < nst) sstore(cur ^ 1);
        __syncthreads();
    }

    const int rowoff = kh * 8;
#pragma unroll
    for (int mi = 0; mi < 2; ++mi) {
#pragma unroll
        for (int ti = 0; ti < 4; ++ti) {
            const int col = n0 + wn * 64 + ti * 16 + ml;
#pragma unroll
            for (int i = 0; i < 8; ++i) {
                float v = acc[mi][ti][i];
                const int row = m0 + wm * 32 + mi * 16 + rowoff + i;
                size_t idx = (size_t)row * N + col;
                if (EPI == 1) { v += bias[col]; v = (v > 20.f) ? v : log1pf(__expf(v)); }
                else if (EPI == 2) { v += resid[idx]; }
                C[idx] = v;
            }
        }
    }
}

// ------------------------------------------------ simple per-wave GEMM (N=32)
template <int NT>
__global__ void k_gemm_wmma_nt(const bf16_t* __restrict__ A,
                               const bf16_t* __restrict__ BT,
                               float* __restrict__ C,
                               int M, int N, int K) {
    const int lane  = threadIdx.x & 31;
    const int gwave = (int)((blockIdx.x * blockDim.x + threadIdx.x) >> 5);
    const int ntn   = N / (16 * NT);
    const int tm    = gwave / ntn;
    const int tn    = gwave - tm * ntn;
    const int mrow  = tm * 16 + (lane & 15);
    const int kh    = lane >> 4;

    v8f acc[NT] = {};
    const bf16_t* arow = A + (size_t)mrow * K;
    for (int k0 = 0; k0 < K; k0 += 32) {
        v8bf alo = *(const v8bf*)(arow + k0 + kh * 8);
        v8bf ahi = *(const v8bf*)(arow + k0 + 16 + kh * 8);
        v16bf af;
#pragma unroll
        for (int j = 0; j < 8; ++j) { af[j] = alo[j]; af[8 + j] = ahi[j]; }
#pragma unroll
        for (int t = 0; t < NT; ++t) {
            const int ncol = (tn * NT + t) * 16 + (lane & 15);
            const bf16_t* brow = BT + (size_t)ncol * K + k0 + kh * 8;
            v8bf blo = *(const v8bf*)(brow);
            v8bf bhi = *(const v8bf*)(brow + 16);
            v16bf bfg;
#pragma unroll
            for (int j = 0; j < 8; ++j) { bfg[j] = blo[j]; bfg[8 + j] = bhi[j]; }
            acc[t] = __builtin_amdgcn_wmma_f32_16x16x32_bf16(
                false, af, false, bfg, (short)0, acc[t], false, false);
        }
    }
    const int m0 = tm * 16 + kh * 8;
    const int nb = lane & 15;
#pragma unroll
    for (int t = 0; t < NT; ++t) {
        const int ncol = (tn * NT + t) * 16 + nb;
#pragma unroll
        for (int i = 0; i < 8; ++i)
            C[(size_t)(m0 + i) * N + ncol] = acc[t][i];
    }
}

// --------------------------------------------- depthwise causal conv + SiLU
__global__ void k_conv_silu(const float* __restrict__ xz,
                            const float* __restrict__ wconv,
                            const float* __restrict__ bconv,
                            float* __restrict__ xin_f,
                            bf16_t* __restrict__ xin_b) {
    long long i = (long long)blockIdx.x * blockDim.x + threadIdx.x;
    if (i >= (long long)M_ROWS * D_INNER) return;
    int c = (int)(i % D_INNER);
    long long row = i / D_INNER;
    int l = (int)(row % SEQ_L);
    float acc = bconv[c];
    const float* base = xz + row * (long long)XZW + c;
#pragma unroll
    for (int j = 0; j < D_CONV; ++j) {
        int ll = l - (D_CONV - 1) + j;
        if (ll >= 0) acc += wconv[c * D_CONV + j] * base[(long long)(ll - l) * XZW];
    }
    float s = acc * sigmoidf_(acc);
    xin_f[i] = s;
    xin_b[i] = (bf16_t)s;
}

// ===================================================== chunked selective scan
__global__ void k_scan_pass1(const float* __restrict__ delta,
                             const float* __restrict__ xin,
                             const float* __restrict__ btct,
                             const float* __restrict__ A_log,
                             float* __restrict__ Hend,   // [B][NCH][16][D_INNER]
                             float* __restrict__ Dts) {  // [B][NCH][D_INNER]
    const int d = blockIdx.x * blockDim.x + threadIdx.x;
    const int c = blockIdx.y;
    const int b = blockIdx.z;
    if (d >= D_INNER) return;
    float Av[D_STATE], h[D_STATE];
#pragma unroll
    for (int n = 0; n < D_STATE; ++n) { Av[n] = -__expf(A_log[n]); h[n] = 0.f; }
    float dts = 0.f;
    const size_t row0 = (size_t)b * SEQ_L + (size_t)c * CLEN;
    for (int l = 0; l < CLEN; ++l) {
        const size_t row = row0 + l;
        float dt = delta[row * D_INNER + d];
        float xt = xin[row * D_INNER + d];
        float dx = dt * xt;
        dts += dt;
        const float* bc = btct + row * (2 * D_STATE);
#pragma unroll
        for (int n = 0; n < D_STATE; ++n)
            h[n] = __expf(dt * Av[n]) * h[n] + dx * bc[n];
    }
    const size_t base = ((size_t)b * NCH + c) * D_STATE * D_INNER;
#pragma unroll
    for (int n = 0; n < D_STATE; ++n) Hend[base + (size_t)n * D_INNER + d] = h[n];
    Dts[((size_t)b * NCH + c) * D_INNER + d] = dts;
}

__global__ void k_scan_pass2(const float* __restrict__ Hend,
                             const float* __restrict__ Dts,
                             const float* __restrict__ A_log,
                             float* __restrict__ Hinit) {
    const int d = blockIdx.x * blockDim.x + threadIdx.x;
    const int b = blockIdx.y;
    if (d >= D_INNER) return;
    float Av[D_STATE], h[D_STATE];
#pragma unroll
    for (int n = 0; n < D_STATE; ++n) { Av[n] = -__expf(A_log[n]); h[n] = 0.f; }
    for (int c = 0; c < NCH; ++c) {
        const size_t base = ((size_t)b * NCH + c) * D_STATE * D_INNER;
#pragma unroll
        for (int n = 0; n < D_STATE; ++n) Hinit[base + (size_t)n * D_INNER + d] = h[n];
        const float dts = Dts[((size_t)b * NCH + c) * D_INNER + d];
#pragma unroll
        for (int n = 0; n < D_STATE; ++n)
            h[n] = __expf(dts * Av[n]) * h[n] + Hend[base + (size_t)n * D_INNER + d];
    }
}

__global__ void k_scan_pass3(const float* __restrict__ delta,
                             const float* __restrict__ xin,
                             const float* __restrict__ btct,
                             const float* __restrict__ xz,
                             const float* __restrict__ A_log,
                             const float* __restrict__ D_skip,
                             const float* __restrict__ Hinit,
                             bf16_t* __restrict__ Ybf) {
    const int d = blockIdx.x * blockDim.x + threadIdx.x;
    const int c = blockIdx.y;
    const int b = blockIdx.z;
    if (d >= D_INNER) return;
    float Av[D_STATE], h[D_STATE];
    const size_t base = ((size_t)b * NCH + c) * D_STATE * D_INNER;
#pragma unroll
    for (int n = 0; n < D_STATE; ++n) {
        Av[n] = -__expf(A_log[n]);
        h[n] = Hinit[base + (size_t)n * D_INNER + d];
    }
    const float Dc = D_skip[d];
    const size_t row0 = (size_t)b * SEQ_L + (size_t)c * CLEN;
    for (int l = 0; l < CLEN; ++l) {
        const size_t row = row0 + l;
        float dt = delta[row * D_INNER + d];
        float xt = xin[row * D_INNER + d];
        float dx = dt * xt;
        const float* bc = btct + row * (2 * D_STATE);
        float y = 0.f;
#pragma unroll
        for (int n = 0; n < D_STATE; ++n) {
            h[n] = __expf(dt * Av[n]) * h[n] + dx * bc[n];
            y = fmaf(h[n], bc[D_STATE + n], y);
        }
        float zv = xz[row * XZW + D_INNER + d];
        float o = (y + xt * Dc) * (zv * sigmoidf_(zv));
        Ybf[row * D_INNER + d] = (bf16_t)o;
    }
}

// ---------------------------------------------------------------- launcher
extern "C" void kernel_launch(void* const* d_in, const int* in_sizes, int n_in,
                              void* d_out, int out_size, void* d_ws, size_t ws_size,
                              hipStream_t stream) {
    const float* x      = (const float*)d_in[0];
    const float* w_in   = (const float*)d_in[1];
    const float* w_conv = (const float*)d_in[2];
    const float* b_conv = (const float*)d_in[3];
    const float* w_dt   = (const float*)d_in[4];
    const float* b_dt   = (const float*)d_in[5];
    const float* w_B    = (const float*)d_in[6];
    const float* w_C    = (const float*)d_in[7];
    const float* A_log  = (const float*)d_in[8];
    const float* D_sk   = (const float*)d_in[9];
    const float* w_out  = (const float*)d_in[10];
    float* out = (float*)d_out;
    (void)in_sizes; (void)n_in; (void)out_size; (void)ws_size;

    char* ws = (char*)d_ws;
    size_t off = 0;
    auto alloc = [&](size_t bytes) -> char* {
        char* p = ws + off;
        off += (bytes + 255) & ~(size_t)255;
        return p;
    };
    bf16_t* xbf    = (bf16_t*)alloc((size_t)M_ROWS * DIM * 2);
    bf16_t* winT   = (bf16_t*)alloc((size_t)XZW * DIM * 2);
    float*  xzf    = (float*) alloc((size_t)M_ROWS * XZW * 4);
    float*  xinf   = (float*) alloc((size_t)M_ROWS * D_INNER * 4);
    bf16_t* xinb   = (bf16_t*)alloc((size_t)M_ROWS * D_INNER * 2);
    bf16_t* wdtT   = (bf16_t*)alloc((size_t)D_INNER * D_INNER * 2);
    float*  deltaf = (float*) alloc((size_t)M_ROWS * D_INNER * 4);
    bf16_t* wbcT   = (bf16_t*)alloc((size_t)2 * D_STATE * D_INNER * 2);
    float*  btct   = (float*) alloc((size_t)M_ROWS * 2 * D_STATE * 4);
    bf16_t* woutT  = (bf16_t*)alloc((size_t)DIM * D_INNER * 2);
    bf16_t* ybf    = (bf16_t*)alloc((size_t)M_ROWS * D_INNER * 2);
    float*  Hend   = (float*) alloc((size_t)B_SZ * NCH * D_STATE * D_INNER * 4);
    float*  Hinit  = (float*) alloc((size_t)B_SZ * NCH * D_STATE * D_INNER * 4);
    float*  Dts    = (float*) alloc((size_t)B_SZ * NCH * D_INNER * 4);

    // 1) precision conversion + weight transposes
    { int n = M_ROWS * DIM;
      k_cvt_bf16<<<(n + 255) / 256, 256, 0, stream>>>(x, xbf, n); }
    { int n = DIM * XZW;
      k_transpose_bf16<<<(n + 255) / 256, 256, 0, stream>>>(w_in, winT, DIM, XZW); }
    { int n = D_INNER * D_INNER;
      k_transpose_bf16<<<(n + 255) / 256, 256, 0, stream>>>(w_dt, wdtT, D_INNER, D_INNER); }
    { int n = D_INNER * D_STATE;
      k_transpose_bf16<<<(n + 255) / 256, 256, 0, stream>>>(w_B, wbcT, D_INNER, D_STATE); }
    { int n = D_INNER * D_STATE;
      k_transpose_bf16<<<(n + 255) / 256, 256, 0, stream>>>(
          w_C, wbcT + (size_t)D_STATE * D_INNER, D_INNER, D_STATE); }
    { int n = D_INNER * DIM;
      k_transpose_bf16<<<(n + 255) / 256, 256, 0, stream>>>(w_out, woutT, D_INNER, DIM); }

#if USE_TDM
#define BIG_GEMM k_gemm_tdm_wmma
#else
#define BIG_GEMM k_gemm_lds_wmma
#endif

    // 2) xz = x @ w_in            M=16384, N=4096, K=1024
    { int blocks = (M_ROWS / BM) * (XZW / BN);
      BIG_GEMM<0><<<blocks, 256, 0, stream>>>(
          xbf, winT, xzf, nullptr, nullptr, M_ROWS, XZW, DIM); }

    // 3) causal depthwise conv + SiLU
    { long long n = (long long)M_ROWS * D_INNER;
      k_conv_silu<<<(unsigned)((n + 255) / 256), 256, 0, stream>>>(
          xzf, w_conv, b_conv, xinf, xinb); }

    // 4) delta = softplus(xin @ w_dt + b_dt)   M=16384, N=2048, K=2048
    { int blocks = (M_ROWS / BM) * (D_INNER / BN);
      BIG_GEMM<1><<<blocks, 256, 0, stream>>>(
          xinb, wdtT, deltaf, b_dt, nullptr, M_ROWS, D_INNER, D_INNER); }

    // 5) [Bt|Ct] = xin @ [w_B|w_C]             M=16384, N=32, K=2048
    { int waves = (M_ROWS / 16) * ((2 * D_STATE) / 32);
      k_gemm_wmma_nt<2><<<waves / 8, 256, 0, stream>>>(
          xinb, wbcT, btct, M_ROWS, 2 * D_STATE, D_INNER); }

    // 6) chunked selective scan
    { dim3 g(D_INNER / 256, NCH, B_SZ);
      k_scan_pass1<<<g, 256, 0, stream>>>(deltaf, xinf, btct, A_log, Hend, Dts); }
    { dim3 g(D_INNER / 256, B_SZ);
      k_scan_pass2<<<g, 256, 0, stream>>>(Hend, Dts, A_log, Hinit); }
    { dim3 g(D_INNER / 256, NCH, B_SZ);
      k_scan_pass3<<<g, 256, 0, stream>>>(deltaf, xinf, btct, xzf, A_log, D_sk, Hinit, ybf); }

    // 7) out = x + y @ w_out                   M=16384, N=1024, K=2048
    { int blocks = (M_ROWS / BM) * (DIM / BN);
      BIG_GEMM<2><<<blocks, 256, 0, stream>>>(
          ybf, woutT, out, nullptr, x, M_ROWS, DIM, D_INNER); }
}
